// MultiHeadAttention_28174985462227
// MI455X (gfx1250) — matrix operand order repaired and checked by reading, not compiled, not run on hardware
//
#include <hip/hip_runtime.h>

typedef __attribute__((ext_vector_type(16))) __bf16   v16bf;
typedef __attribute__((ext_vector_type(8)))  float    v8f;
typedef __attribute__((ext_vector_type(4)))  unsigned uint4v;

union FragAB { v16bf bf; uint4v q[2]; };

__device__ __forceinline__ unsigned short f2bf(float f) {
  unsigned u = __builtin_bit_cast(unsigned, f);
  u += 0x7FFFu + ((u >> 16) & 1u);   // round-to-nearest-even
  return (unsigned short)(u >> 16);
}

#define WMMA_BF16(A_, B_, C_) \
  __builtin_amdgcn_wmma_f32_16x16x32_bf16(false, (A_), false, (B_), (short)0, (C_), false, false)

// ------------------------------------------------------ TDM (CDNA5 tensor DMA)
#if __has_builtin(__builtin_amdgcn_tensor_load_to_lds)
#define HAVE_TDM 1
typedef __attribute__((ext_vector_type(4))) unsigned u32x4;
typedef __attribute__((ext_vector_type(8))) int      i32x8;
typedef __attribute__((ext_vector_type(4))) int      i32x4;

// DMA a 64-row x 64-byte (16-DWORD) tile of the bf16 A matrix (row stride
// 2048B = 512 DWORDs) into LDS, inserting 16B padding after every 64B so the
// LDS layout matches As[64][40] (80B row stride). Wave-level op; EXEC ignored.
__device__ __forceinline__ void tdm_load_tile_a(const unsigned short* gsrc,
                                                unsigned lds_off) {
  unsigned long long ga = (unsigned long long)gsrc;
  u32x4 g0;
  g0[0] = 1u;                                               // count=1 valid desc
  g0[1] = lds_off;                                          // lds_addr (bytes)
  g0[2] = (unsigned)(ga & 0xFFFFFFFFull);                   // global_addr lo
  g0[3] = (unsigned)((ga >> 32) & 0x1FFFFFFull) | (2u << 30); // addr hi | type=2
  i32x8 g1;
  g1[0] = (2 << 16)        // data_size = 4B
        | (1 << 20)        // pad_enable
        | (3 << 22)        // pad_interval: 16 DWORDs (64B)
        | (3 << 25);       // pad_amount:   4 DWORDs (16B)
  g1[1] = (int)(512u << 16);   // tensor_dim0 = 512 DWORDs (lo16 in [31:16])
  g1[2] = (int)(4096u << 16);  // dim0 hi=0 | tensor_dim1 lo16 = 4096 rows
  g1[3] = (int)(16u << 16);    // dim1 hi=0 | tile_dim0 = 16 DWORDs
  g1[4] = 64;                  // tile_dim1 = 64 rows, tile_dim2 = 0 (2D)
  g1[5] = 512;                 // tensor_dim0_stride = 512 DWORDs
  g1[6] = 0;
  g1[7] = 0;
  i32x4 z4 = {0, 0, 0, 0};     // groups 2/3 unused (2D tensor)
  i32x8 z8 = {0, 0, 0, 0, 0, 0, 0, 0};
  __builtin_amdgcn_tensor_load_to_lds(g0, g1, z4, z4, z8, 0);
}
#else
#define HAVE_TDM 0
#endif

// ---------------------------------------------------------------- convert
__global__ void cvt_f32_bf16(const float* __restrict__ s,
                             unsigned short* __restrict__ d, int n) {
  int i = blockIdx.x * blockDim.x + threadIdx.x;
  if (i < n) d[i] = f2bf(s[i]);
}

// ---------------------------------------------------------------- GEMM
// Y = A(bf16,[4096x1024]) @ B(bf16,[1024x1024]), f32 accum.
// mode 0: out = bf16 head-major Q/K  [((b*16+h)*2048+l)*64+d], *scale
// mode 2: out = bf16 transposed V    [((b*16+h)*64+d)*2048+l]
// mode 1: out = f32 row-major + bias [row*1024+col]
__global__ __launch_bounds__(256) void gemm_bf16_kernel(
    const unsigned short* __restrict__ A,
    const unsigned short* __restrict__ Bw,
    void* __restrict__ outp,
    const float* __restrict__ bias,
    float scale, int mode) {
  constexpr int Kd = 1024, Nd = 1024;
  __shared__ unsigned short As[64][40];
  __shared__ unsigned short Bs[64][40];
  const int t = threadIdx.x;
  const int lane = t & 31, w = t >> 5;
  const int hi = lane >> 4, ln = lane & 15;
  const int rowBase = blockIdx.x * 64;
  const int colBase = blockIdx.y * 64;
  const int mi0 = w >> 2, ni = w & 3, mi1 = mi0 + 2;

  v8f acc0 = {}, acc1 = {};

  const int brow = t >> 3, bseg = (t & 7) * 8;
#if HAVE_TDM
  const unsigned ldsA = (unsigned)(unsigned long long)(&As[0][0]);
#else
  const int arow = t >> 2, aseg = (t & 3) * 8;
#endif

  for (int k0 = 0; k0 < Kd; k0 += 32) {
    uint4v bv = *reinterpret_cast<const uint4v*>(&Bw[(k0 + brow) * Nd + colBase + bseg]);
#if !HAVE_TDM
    uint4v av = *reinterpret_cast<const uint4v*>(&A[(rowBase + arow) * Kd + k0 + aseg]);
#endif
    __syncthreads();                       // previous iteration's LDS reads done
#if HAVE_TDM
    if (w == 0) tdm_load_tile_a(&A[(size_t)rowBase * Kd + k0], ldsA);
#else
    *reinterpret_cast<uint4v*>(&As[arow][aseg]) = av;
#endif
    union { uint4v v; unsigned short s[8]; } bu; bu.v = bv;
#pragma unroll
    for (int e = 0; e < 8; ++e) Bs[bseg + e][brow] = bu.s[e];
#if HAVE_TDM
    if (w == 0) __builtin_amdgcn_s_wait_tensorcnt(0);
#endif
    __syncthreads();                       // tile (TDM + transpose) visible

    FragAB a0, a1, b;
    a0.q[0] = *(const uint4v*)&As[mi0 * 16 + ln][hi * 8];
    a0.q[1] = *(const uint4v*)&As[mi0 * 16 + ln][16 + hi * 8];
    a1.q[0] = *(const uint4v*)&As[mi1 * 16 + ln][hi * 8];
    a1.q[1] = *(const uint4v*)&As[mi1 * 16 + ln][16 + hi * 8];
    b.q[0]  = *(const uint4v*)&Bs[ni * 16 + ln][hi * 8];
    b.q[1]  = *(const uint4v*)&Bs[ni * 16 + ln][16 + hi * 8];
    acc0 = WMMA_BF16(a0.bf, b.bf, acc0);
    acc1 = WMMA_BF16(a1.bf, b.bf, acc1);
  }

#pragma unroll
  for (int half = 0; half < 2; ++half) {
    const v8f acc = half ? acc1 : acc0;
    const int mi = half ? mi1 : mi0;
#pragma unroll
    for (int r = 0; r < 8; ++r) {
      const int row = rowBase + mi * 16 + r + 8 * hi;
      const int col = colBase + ni * 16 + ln;
      const float v = acc[r];
      if (mode == 1) {
        ((float*)outp)[(size_t)row * Nd + col] = v + bias[col];
      } else {
        const int b_ = row >> 11, l = row & 2047;
        const int h_ = col >> 6,  d = col & 63;
        const unsigned short bb = f2bf(v * scale);
        if (mode == 0)
          ((unsigned short*)outp)[(((size_t)(b_ * 16 + h_) * 2048) + l) * 64 + d] = bb;
        else
          ((unsigned short*)outp)[(((size_t)(b_ * 16 + h_) * 64) + d) * 2048 + l] = bb;
      }
    }
  }
}

// ---------------------------------------------------------------- attention
// One wave per (bh, 16-query tile). Causal flash attention + relative bias.
__global__ __launch_bounds__(128) void attn_kernel(
    const unsigned short* __restrict__ Qb,   // [bh][l][64] bf16 (prescaled 1/8)
    const unsigned short* __restrict__ Kb,   // [bh][l][64] bf16
    const unsigned short* __restrict__ Vt,   // [bh][64][l] bf16 (transposed)
    const unsigned short* __restrict__ relb, // [4095][64]  bf16
    unsigned short* __restrict__ ctxb) {     // [b*2048+l][1024] bf16
  constexpr int L = 2048, HD = 64;
  __shared__ float          Pls[4][16][48];
  __shared__ unsigned short Prob[4][16][32];
  const int lane = threadIdx.x & 31, w = threadIdx.x >> 5;
  const int hi = lane >> 4, ln = lane & 15;
  const int l0 = blockIdx.x * 16;
  const int bh = blockIdx.y * 4 + w;
  const unsigned short* Qp = Qb + (size_t)bh * L * HD;
  const unsigned short* Kp = Kb + (size_t)bh * L * HD;
  const unsigned short* Vp = Vt + (size_t)bh * HD * L;

  FragAB aq[2];
#pragma unroll
  for (int c = 0; c < 2; ++c) {
    const unsigned short* qr = &Qp[(size_t)(l0 + ln) * HD + c * 32];
    aq[c].q[0] = *(const uint4v*)&qr[hi * 8];
    aq[c].q[1] = *(const uint4v*)&qr[16 + hi * 8];
  }

  v8f o[4] = {};
  float rmax[8], rsum[8];
#pragma unroll
  for (int r = 0; r < 8; ++r) { rmax[r] = -3.0e38f; rsum[r] = 0.f; }

  const int nkb = (l0 + 47) >> 5;
  for (int kb = 0; kb < nkb; ++kb) {
    const int r0 = kb * 32;

    // content scores: two 16x16 tiles over 32 keys
    v8f s0 = {}, s1 = {};
#pragma unroll
    for (int c = 0; c < 2; ++c) {
      FragAB bk;
      const unsigned short* kp0 = &Kp[(size_t)(r0 + ln) * HD + c * 32 + hi * 8];
      bk.q[0] = *(const uint4v*)&kp0[0];
      bk.q[1] = *(const uint4v*)&kp0[16];
      s0 = WMMA_BF16(aq[c].bf, bk.bf, s0);
      const unsigned short* kp1 = &Kp[(size_t)(r0 + 16 + ln) * HD + c * 32 + hi * 8];
      bk.q[0] = *(const uint4v*)&kp1[0];
      bk.q[1] = *(const uint4v*)&kp1[16];
      s1 = WMMA_BF16(aq[c].bf, bk.bf, s1);
    }

    // relative-bias panel P[i][u] = q[l0+i] . rel_emb[tbase+u], u in [0,48)
    const int tbase = 2032 + r0 - l0;
    v8f p0 = {}, p1 = {}, p2 = {};
#pragma unroll
    for (int ns = 0; ns < 3; ++ns) {
#pragma unroll
      for (int c = 0; c < 2; ++c) {
        FragAB br;
        const unsigned short* rp = &relb[(size_t)(tbase + ns * 16 + ln) * HD + c * 32 + hi * 8];
        br.q[0] = *(const uint4v*)&rp[0];
        br.q[1] = *(const uint4v*)&rp[16];
        if (ns == 0) p0 = WMMA_BF16(aq[c].bf, br.bf, p0);
        else if (ns == 1) p1 = WMMA_BF16(aq[c].bf, br.bf, p1);
        else p2 = WMMA_BF16(aq[c].bf, br.bf, p2);
      }
    }
#pragma unroll
    for (int r = 0; r < 8; ++r) {
      Pls[w][r + 8 * hi][ln]      = p0[r];
      Pls[w][r + 8 * hi][16 + ln] = p1[r];
      Pls[w][r + 8 * hi][32 + ln] = p2[r];
    }
    asm volatile("s_wait_dscnt 0" ::: "memory");

    // bias gather (skew), causal mask, online softmax
#pragma unroll
    for (int r = 0; r < 8; ++r) {
      const int i = r + 8 * hi;
      float a = s0[r] + Pls[w][i][ln - i + 15];
      float b = s1[r] + Pls[w][i][ln + 31 - i];
      if (r0 + ln > l0 + i)      a = -3.0e38f;
      if (r0 + 16 + ln > l0 + i) b = -3.0e38f;
      float mx = fmaxf(a, b);
      mx = fmaxf(mx, __shfl_xor(mx, 1));
      mx = fmaxf(mx, __shfl_xor(mx, 2));
      mx = fmaxf(mx, __shfl_xor(mx, 4));
      mx = fmaxf(mx, __shfl_xor(mx, 8));
      const float mt = fmaxf(rmax[r], mx);
      const float corr = __expf(rmax[r] - mt);
      rmax[r] = mt;
      const float pa = __expf(a - mt), pb = __expf(b - mt);
      float bs = pa + pb;
      bs += __shfl_xor(bs, 1);
      bs += __shfl_xor(bs, 2);
      bs += __shfl_xor(bs, 4);
      bs += __shfl_xor(bs, 8);
      rsum[r] = rsum[r] * corr + bs;
#pragma unroll
      for (int dsub = 0; dsub < 4; ++dsub) o[dsub][r] *= corr;
      Prob[w][i][ln]      = f2bf(pa);
      Prob[w][i][16 + ln] = f2bf(pb);
    }
    asm volatile("s_wait_dscnt 0" ::: "memory");

    // probs (A frag) x V block (B frags, contiguous thanks to V^T layout)
    FragAB ap;
    ap.q[0] = *(const uint4v*)&Prob[w][ln][hi * 8];
    ap.q[1] = *(const uint4v*)&Prob[w][ln][16 + hi * 8];
#pragma unroll
    for (int dsub = 0; dsub < 4; ++dsub) {
      FragAB bv;
      const unsigned short* vp = &Vp[(size_t)(dsub * 16 + ln) * L + r0 + hi * 8];
      bv.q[0] = *(const uint4v*)&vp[0];
      bv.q[1] = *(const uint4v*)&vp[16];
      o[dsub] = WMMA_BF16(ap.bf, bv.bf, o[dsub]);
    }
  }

  // write ctx (bf16) in GEMM-input layout [b*L+l][h*64+d]
  const int b_ = bh >> 4, h_ = bh & 15;
#pragma unroll
  for (int r = 0; r < 8; ++r) {
    const int l = l0 + r + 8 * hi;
    const float inv = 1.0f / rsum[r];
#pragma unroll
    for (int dsub = 0; dsub < 4; ++dsub) {
      const int d = dsub * 16 + ln;
      ctxb[((size_t)(b_ * L + l)) * 1024 + h_ * 64 + d] = f2bf(o[dsub][r] * inv);
    }
  }
}

// ---------------------------------------------------------------- launch
extern "C" void kernel_launch(void* const* d_in, const int* in_sizes, int n_in,
                              void* d_out, int out_size, void* d_ws, size_t ws_size,
                              hipStream_t stream) {
  (void)in_sizes; (void)n_in; (void)out_size; (void)ws_size;
  constexpr int B = 2, L = 2048, D = 1024, H = 16, HD = 64;
  const float* x    = (const float*)d_in[0];
  const float* W_q  = (const float*)d_in[1];
  const float* W_k  = (const float*)d_in[2];
  const float* W_v  = (const float*)d_in[3];
  const float* W_o  = (const float*)d_in[4];
  const float* b_o  = (const float*)d_in[5];
  const float* rel  = (const float*)d_in[6];

  unsigned short* xb   = (unsigned short*)d_ws;               // 4096*1024
  unsigned short* wqb  = xb   + (size_t)B * L * D;
  unsigned short* wkb  = wqb  + (size_t)D * D;
  unsigned short* wvb  = wkb  + (size_t)D * D;
  unsigned short* wob  = wvb  + (size_t)D * D;
  unsigned short* relb = wob  + (size_t)D * D;                // 4095*64 (pad 4096*64)
  unsigned short* qb   = relb + (size_t)4096 * HD;
  unsigned short* kb   = qb   + (size_t)B * H * L * HD;
  unsigned short* vtb  = kb   + (size_t)B * H * L * HD;
  unsigned short* ctxb = vtb  + (size_t)B * H * L * HD;       // 4096*1024

  const int nx = B * L * D, nw = D * D, nr = (2 * L - 1) * HD;
  cvt_f32_bf16<<<(nx + 255) / 256, 256, 0, stream>>>(x, xb, nx);
  cvt_f32_bf16<<<(nw + 255) / 256, 256, 0, stream>>>(W_q, wqb, nw);
  cvt_f32_bf16<<<(nw + 255) / 256, 256, 0, stream>>>(W_k, wkb, nw);
  cvt_f32_bf16<<<(nw + 255) / 256, 256, 0, stream>>>(W_v, wvb, nw);
  cvt_f32_bf16<<<(nw + 255) / 256, 256, 0, stream>>>(W_o, wob, nw);
  cvt_f32_bf16<<<(nr + 255) / 256, 256, 0, stream>>>(rel, relb, nr);

  dim3 gg(64, 16);
  gemm_bf16_kernel<<<gg, 256, 0, stream>>>(xb, wqb, qb,  nullptr, 0.125f, 0); // Q * 1/sqrt(64)
  gemm_bf16_kernel<<<gg, 256, 0, stream>>>(xb, wkb, kb,  nullptr, 1.0f,   0); // K
  gemm_bf16_kernel<<<gg, 256, 0, stream>>>(xb, wvb, vtb, nullptr, 1.0f,   2); // V (transposed)

  attn_kernel<<<dim3(L / 16, (B * H) / 4), 128, 0, stream>>>(qb, kb, vtb, relb, ctxb);

  gemm_bf16_kernel<<<gg, 256, 0, stream>>>(ctxb, wob, d_out, b_o, 1.0f, 1);   // ctx @ W_o + b_o
}
